// RecurrentResonanceModel_52183852646985
// MI455X (gfx1250) — matrix operand order repaired and checked by reading, not compiled, not run on hardware
//
#include <hip/hip_runtime.h>
#include <hip/hip_bf16.h>
#include <math.h>

typedef __attribute__((ext_vector_type(16))) _Float16 v16h;
typedef __attribute__((ext_vector_type(8)))  float    v8f;

#define B_    2
#define E_    512
#define L_    128
#define CH_   256
#define NC_   257
#define NF_   128
#define STEP_ 256
#define RS_   32768
#define NKT   17          // K tiles: ceil(2*257/32) -> K padded to 544
#define NNT   32          // N tiles: 512/16
#define BE_   (B_*E_)     // 1024

static __device__ __forceinline__ float sigmoidf_(float x) {
    return 1.0f / (1.0f + __expf(-x));
}

// ---------------- zero the output (OLA accumulates atomically) ----------------
__global__ void k_zero(float4* __restrict__ out) {
    out[(size_t)blockIdx.x * 256 + threadIdx.x] = make_float4(0.f, 0.f, 0.f, 0.f);
}

// ---------------- constant irfft basis, pre-swizzled to WMMA B-fragment layout
// Bpack[nt][kt][lane][e]  (f16).  N = nt*16 + (lane&15)
// K packing (16-bit B operand): lanes 0-15 hold K=kt*32+e, lanes 16-31 hold K=kt*32+16+e
__global__ void k_basis(_Float16* __restrict__ Bpack) {
    int idx = blockIdx.x * 256 + threadIdx.x;          // NNT*NKT*32*16 = 278528
    if (idx >= NNT * NKT * 32 * 16) return;
    int e = idx & 15; int r = idx >> 4;
    int lane = r & 31; r >>= 5;
    int kt = r % NKT;  int nt = r / NKT;
    int N  = nt * 16 + (lane & 15);
    int kk = ((lane < 16) ? 0 : 16) + e;
    int kg = kt * 32 + kk;                              // interleaved: even=Re row, odd=Im row
    float val = 0.0f;
    if (kg < 2 * NC_) {
        int n = kg >> 1; int isIm = kg & 1;
        const float scale = 0.04419417382415922f;       // 1/sqrt(512), ortho norm
        if (n == 0)        val = isIm ? 0.0f : scale;
        else if (n == 256) val = isIm ? 0.0f : ((N & 1) ? -scale : scale);
        else {
            int m = (N * n) & 511;                      // exact mod-512 angle reduction
            float ang = (float)m * (float)(M_PI / 256.0);
            val = isIm ? (-2.0f * scale * __sinf(ang)) : (2.0f * scale * __cosf(ang));
        }
    }
    Bpack[idx] = (_Float16)val;
}

// ---------------- 3 MLP stacks (f32, precision-sensitive), one WG per (s,b,e) row
__global__ void __launch_bounds__(256) k_mlp(
        const float* __restrict__ latents, const float* __restrict__ Win,
        const float* __restrict__ b_in,    const float* __restrict__ Wh,
        const float* __restrict__ b_h,     const float* __restrict__ Wout,
        const float* __restrict__ b_out,   float* __restrict__ raw) {
    __shared__ float xin[L_];
    __shared__ float hbuf[CH_];
    __shared__ float red[256];
    int tid = threadIdx.x;
    int wg  = blockIdx.x;                  // 3*1024
    int s   = wg >> 10;
    int be  = wg & 1023;

    if (tid < L_) xin[tid] = latents[be * L_ + tid];
    __syncthreads();

    // input layer: 128 -> 256
    float h = b_in[s * CH_ + tid];
    const float* W0 = Win + s * L_ * CH_ + tid;
    #pragma unroll 4
    for (int l = 0; l < L_; ++l) h += xin[l] * W0[l * CH_];

    // (lrelu + LN [+ hidden GEMM]) x3, last pass without GEMM
    for (int k = 0; k < 3; ++k) {
        h = (h > 0.f) ? h : 0.2f * h;
        red[tid] = h; __syncthreads();
        for (int off = 128; off > 0; off >>= 1) {
            if (tid < off) red[tid] += red[tid + off];
            __syncthreads();
        }
        float m = red[0] * (1.0f / CH_); __syncthreads();
        float d = h - m;
        red[tid] = d * d; __syncthreads();
        for (int off = 128; off > 0; off >>= 1) {
            if (tid < off) red[tid] += red[tid + off];
            __syncthreads();
        }
        float v = red[0] * (1.0f / CH_); __syncthreads();
        h = d * rsqrtf(v + 1e-5f);

        if (k < 2) {
            hbuf[tid] = h; __syncthreads();
            float acc = b_h[(s * 2 + k) * CH_ + tid];
            const float* Wk = Wh + ((s * 2 + k) * CH_) * CH_ + tid;
            #pragma unroll 4
            for (int c = 0; c < CH_; ++c) acc += hbuf[c] * Wk[c * CH_];
            __syncthreads();
            h = acc;
        }
    }

    // output layer: 256 -> 257
    hbuf[tid] = h; __syncthreads();
    for (int n = tid; n < NC_; n += 256) {
        float acc = b_out[s * NC_ + n];
        const float* Wo = Wout + s * CH_ * NC_ + n;
        #pragma unroll 4
        for (int c = 0; c < CH_; ++c) acc += hbuf[c] * Wo[c * NC_];
        raw[(s * BE_ + be) * NC_ + n] = acc;
    }
}

// ---------------- fused synth: frames -> WMMA irfft GEMM -> windowed OLA
// one workgroup (8 waves) per (b,e); owns all 128 frames
__global__ void __launch_bounds__(256, 1) k_synth(
        const float* __restrict__ raw,       const float* __restrict__ phase0_u,
        const float* __restrict__ noise_u,   const _Float16* __restrict__ Bpack,
        float* __restrict__ out) {
    // A in WMMA A-fragment layout: [mtile=8][kt=17][lane=32][e=16] f16 (139264 B)
    __shared__ __align__(32) _Float16 Apack[8 * NKT * 32 * 16];
    __shared__ __align__(32) _Float16 Bstage[NKT * 32 * 16];     // 17408 B
    int tid = threadIdx.x;
    int be  = blockIdx.x;
    int b   = be >> 9, e = be & 511;

    // zero Apack (covers K padding 514..543)
    {
        float4* p = (float4*)Apack;             // 8704 float4
        float4 z = make_float4(0.f, 0.f, 0.f, 0.f);
        for (int i = tid; i < 8704; i += 256) p[i] = z;
    }
    __syncthreads();

    // ---- phase 1: per-n sequential recurrence over t, scatter Re/Im into Apack
    for (int n = tid; n < NC_; n += 256) {
        float init = raw[be * NC_ + n];
        float r1   = raw[1 * BE_ * NC_ + be * NC_ + n];
        float r2   = raw[2 * BE_ * NC_ + be * NC_ + n];
        float res  = 0.5f + sigmoidf_(r1) * 0.4995f;      // BASE + sig*FACTOR
        float dith = sigmoidf_(r2);
        float gd   = (float)(M_PI / 256.0) * (float)n;    // group delay
        float ph   = (phase0_u[be * NC_ + n] * 2.0f - 1.0f) * (float)M_PI;
        float mag  = init;
        const float inv2pi = 0.15915494309189535f;
        const float twopi  = 6.283185307179586f;
        for (int t = 0; t < NF_; ++t) {
            if (t > 0) {
                float nz = (noise_u[((b * NF_ + t) * E_ + e) * NC_ + n] * 2.0f - 1.0f)
                           * (float)M_PI;                 // noise[:,0] is zeroed in ref
                ph += gd + dith * nz;
                ph -= twopi * rintf(ph * inv2pi);         // keep wrapped for v_sin/v_cos
            }
            float s_, c_;
            __sincosf(ph, &s_, &c_);
            float re = mag * c_, im = mag * s_;
            int mt = t >> 4, M = t & 15;
            // scatter into A-fragment layout (contiguous: lanes hi hold K=16..31)
            #pragma unroll
            for (int q = 0; q < 2; ++q) {
                int kg = 2 * n + q;
                int kt = kg >> 5, kk = kg & 31;
                int laneoff = (kk & 16) ? 16 : 0;
                int ee = kk & 15;
                Apack[(((mt * NKT) + kt) * 32 + (M + laneoff)) * 16 + ee] =
                    (_Float16)(q ? im : re);
            }
            mag *= res;
        }
    }
    __syncthreads();

    // ---- phase 2: WMMA GEMM  X[128 x 512] = A[128 x 544] * B[544 x 512]
    int w = tid >> 5, lane = tid & 31;
    v16h a[NKT];
    #pragma unroll
    for (int kt = 0; kt < NKT; ++kt)
        a[kt] = *(const v16h*)&Apack[((w * NKT + kt) * 32 + lane) * 16];

    for (int nt = 0; nt < NNT; ++nt) {
        __syncthreads();   // previous Bstage readers done
        {
            const float4* src = (const float4*)(Bpack + (size_t)nt * (NKT * 32 * 16));
            float4* dst = (float4*)Bstage;
            for (int i = tid; i < 1088; i += 256) dst[i] = src[i];
        }
        __syncthreads();

        v8f acc = {0.f, 0.f, 0.f, 0.f, 0.f, 0.f, 0.f, 0.f};
        #pragma unroll
        for (int kt = 0; kt < NKT; ++kt) {
            v16h bf = *(const v16h*)&Bstage[(kt * 32 + lane) * 16];
            acc = __builtin_amdgcn_wmma_f32_16x16x32_f16(
                false, a[kt], false, bf, (short)0, acc, false, false);
        }

        // epilogue: Hann window + overlap-add (frame t sample j -> out[t*256 + j])
        int j = nt * 16 + (lane & 15);
        float win = 0.5f - 0.5f * __cosf((float)j * (float)(M_PI / 256.0));
        #pragma unroll
        for (int v = 0; v < 8; ++v) {
            int t = w * 16 + ((lane < 16) ? v : (v + 8));
            int pos = t * STEP_ + j;
            if (pos < RS_) {
                __hip_atomic_fetch_add(&out[(size_t)be * RS_ + pos], win * acc[v],
                                       __ATOMIC_RELAXED, __HIP_MEMORY_SCOPE_AGENT);
            }
        }
    }
}

extern "C" void kernel_launch(void* const* d_in, const int* in_sizes, int n_in,
                              void* d_out, int out_size, void* d_ws, size_t ws_size,
                              hipStream_t stream) {
    const float* latents  = (const float*)d_in[0];
    const float* phase0_u = (const float*)d_in[1];
    const float* noise_u  = (const float*)d_in[2];
    const float* Win      = (const float*)d_in[3];
    const float* b_in     = (const float*)d_in[4];
    const float* Wh       = (const float*)d_in[5];
    const float* b_h      = (const float*)d_in[6];
    const float* Wout     = (const float*)d_in[7];
    const float* b_out    = (const float*)d_in[8];
    float* out = (float*)d_out;

    // workspace: raw3 [3*1024*257] f32, then Bpack [32*17*32*16] f16 (~3.7 MB total)
    float*     raw   = (float*)d_ws;
    _Float16*  Bpack = (_Float16*)((char*)d_ws + (size_t)3 * BE_ * NC_ * sizeof(float));

    k_zero <<<(size_t)BE_ * RS_ / (4 * 256), 256, 0, stream>>>((float4*)out);
    k_basis<<<(NNT * NKT * 32 * 16 + 255) / 256, 256, 0, stream>>>(Bpack);
    k_mlp  <<<3 * BE_, 256, 0, stream>>>(latents, Win, b_in, Wh, b_h, Wout, b_out, raw);
    k_synth<<<BE_, 256, 0, stream>>>(raw, phase0_u, noise_u, Bpack, out);
}
